// SRINR_1400159339023
// MI455X (gfx1250) — matrix operand order repaired and checked by reading, not compiled, not run on hardware
//
#include <hip/hip_runtime.h>
#include <hip/hip_bf16.h>
#include <cmath>

// ---------------------------------------------------------------------------
// Types for CDNA5 WMMA (wave32)
// ---------------------------------------------------------------------------
typedef _Float16 f16;
typedef __attribute__((ext_vector_type(16))) _Float16 v16h;
typedef __attribute__((ext_vector_type(8)))  _Float16 v8h;
typedef __attribute__((ext_vector_type(8)))  float    v8f;

#define NLEV   12
#define FPL    8
#define Bv     2
#define Nv     1024
#define PTS    (Bv * Nv)          // 2048
#define HIDDEN 256
#define INDIM  354                // 2 + 96 + 256
#define KP0    384                // INDIM padded to mult of 32
#define MAXKP  384

__device__ __forceinline__ float gelu_exact(float x) {
    return 0.5f * x * (1.0f + erff(x * 0.70710678118654752f));
}

// ---------------------------------------------------------------------------
// 1) Multi-resolution hash encoding: (2048 pts) x 12 levels -> feats[l][p][8]
// ---------------------------------------------------------------------------
struct HashArgs {
    const float* tables[NLEV];
    long long    hsz[NLEV];
    int          res[NLEV];
};

__global__ void hash_encode_kernel(const float* __restrict__ coords,
                                   float* __restrict__ feats, HashArgs ha) {
    int t = blockIdx.x * blockDim.x + threadIdx.x;
    if (t >= NLEV * PTS) return;
    int l = t / PTS, p = t % PTS;
    float cx = fminf(fmaxf(coords[p * 2 + 0], 0.f), 1.f);
    float cy = fminf(fmaxf(coords[p * 2 + 1], 0.f), 1.f);
    int res = ha.res[l];
    long long hsz = ha.hsz[l];
    const float* tab = ha.tables[l];
    float sx = cx * (float)res, sy = cy * (float)res;
    long long bx = (long long)floorf(sx), by = (long long)floorf(sy);
    float dx = sx - (float)bx, dy = sy - (float)by;
    float w[4] = {(1.f-dx)*(1.f-dy), (1.f-dx)*dy, dx*(1.f-dy), dx*dy};
    const int ox[4] = {0,0,1,1}, oy[4] = {0,1,0,1};
    float acc[FPL];
    #pragma unroll
    for (int d = 0; d < FPL; ++d) acc[d] = 0.f;
    #pragma unroll
    for (int c = 0; c < 4; ++c) {
        long long x = bx + ox[c]; if (x < 0) x = 0; if (x > res-1) x = res-1;
        long long y = by + oy[c]; if (y < 0) y = 0; if (y > res-1) y = res-1;
        long long h = (x ^ (y * 2654435761LL)) % hsz;
        const float* e = tab + h * FPL;
        #pragma unroll
        for (int d = 0; d < FPL; ++d) acc[d] += w[c] * e[d];
    }
    float* o = feats + ((size_t)l * PTS + p) * FPL;
    #pragma unroll
    for (int d = 0; d < FPL; ++d) o[d] = acc[d];
}

// ---------------------------------------------------------------------------
// 2) Attention blocks (D=8, 4 heads, head-dim 2). VALU path: K too skinny for
//    the matrix pipe. qkv precompute, then per-point online-softmax attention.
// ---------------------------------------------------------------------------
struct AttnP {
    const float *ff_b1, *ff_b2, *ff_w1, *ff_w2, *in_b, *in_w;
    const float *ln1_b, *ln1_g, *ln2_b, *ln2_g, *out_b, *out_w;
};

__global__ void qkv_kernel(const float* __restrict__ x, int xs,
                           const float* __restrict__ ctx, int cs,
                           float* __restrict__ q, float* __restrict__ k,
                           float* __restrict__ v, AttnP p) {
    int pt = blockIdx.x * blockDim.x + threadIdx.x;
    if (pt >= PTS) return;
    float xi[8], ci[8];
    #pragma unroll
    for (int d = 0; d < 8; ++d) { xi[d] = x[(size_t)pt*xs + d]; ci[d] = ctx[(size_t)pt*cs + d]; }
    #pragma unroll
    for (int d = 0; d < 8; ++d) {
        float aq = p.in_b[d], ak = p.in_b[8 + d], av = p.in_b[16 + d];
        #pragma unroll
        for (int e = 0; e < 8; ++e) {
            aq += xi[e] * p.in_w[d * 8 + e];
            ak += ci[e] * p.in_w[(8 + d) * 8 + e];
            av += ci[e] * p.in_w[(16 + d) * 8 + e];
        }
        q[pt*8+d] = aq; k[pt*8+d] = ak; v[pt*8+d] = av;
    }
}

__device__ __forceinline__ void ln8(float* x, const float* g, const float* b) {
    float m = 0.f;
    #pragma unroll
    for (int d = 0; d < 8; ++d) m += x[d];
    m *= 0.125f;
    float vv = 0.f;
    #pragma unroll
    for (int d = 0; d < 8; ++d) { float t = x[d] - m; vv += t * t; }
    vv *= 0.125f;
    float inv = rsqrtf(vv + 1e-5f);
    #pragma unroll
    for (int d = 0; d < 8; ++d) x[d] = (x[d] - m) * inv * g[d] + b[d];
}

__global__ void attn_point_kernel(const float* __restrict__ x, int xs,
                                  const float* __restrict__ q,
                                  const float* __restrict__ k,
                                  const float* __restrict__ v,
                                  float* __restrict__ out, int os, AttnP p) {
    int pt = blockIdx.x * blockDim.x + threadIdx.x;
    if (pt >= PTS) return;
    int b = pt >> 10;
    const float* kb = k + (size_t)b * Nv * 8;
    const float* vb = v + (size_t)b * Nv * 8;
    float qi[8];
    #pragma unroll
    for (int d = 0; d < 8; ++d) qi[d] = q[pt * 8 + d];
    float mx[4], sm[4], a0[4], a1[4];
    #pragma unroll
    for (int h = 0; h < 4; ++h) { mx[h] = -1e30f; sm[h] = 0.f; a0[h] = 0.f; a1[h] = 0.f; }
    const float scale = 0.70710678118654752f; // 1/sqrt(head_dim=2)
    for (int j = 0; j < Nv; ++j) {
        const float* kj = kb + j * 8;
        const float* vj = vb + j * 8;
        if ((j & 63) == 0 && j + 64 < Nv) {
            // gfx1250 global_prefetch: stream K/V 64 rows ahead
            __builtin_prefetch(kj + 64 * 8, 0, 1);
            __builtin_prefetch(vj + 64 * 8, 0, 1);
        }
        #pragma unroll
        for (int h = 0; h < 4; ++h) {
            float sc = (qi[2*h]*kj[2*h] + qi[2*h+1]*kj[2*h+1]) * scale;
            if (sc > mx[h]) {
                float r = expf(mx[h] - sc);
                sm[h] *= r; a0[h] *= r; a1[h] *= r; mx[h] = sc;
            }
            float e = expf(sc - mx[h]);
            sm[h] += e; a0[h] += e * vj[2*h]; a1[h] += e * vj[2*h+1];
        }
    }
    float o[8];
    #pragma unroll
    for (int h = 0; h < 4; ++h) { o[2*h] = a0[h] / sm[h]; o[2*h+1] = a1[h] / sm[h]; }
    float xn[8];
    #pragma unroll
    for (int d = 0; d < 8; ++d) {
        float t = p.out_b[d];
        #pragma unroll
        for (int e = 0; e < 8; ++e) t += o[e] * p.out_w[d * 8 + e];
        xn[d] = x[(size_t)pt*xs + d] + t;
    }
    ln8(xn, p.ln1_g, p.ln1_b);
    float h16[16];
    #pragma unroll
    for (int i = 0; i < 16; ++i) {
        float t = p.ff_b1[i];
        #pragma unroll
        for (int e = 0; e < 8; ++e) t += xn[e] * p.ff_w1[i * 8 + e];
        h16[i] = gelu_exact(t);
    }
    float y[8];
    #pragma unroll
    for (int d = 0; d < 8; ++d) {
        float t = p.ff_b2[d];
        #pragma unroll
        for (int i = 0; i < 16; ++i) t += h16[i] * p.ff_w2[d * 16 + i];
        y[d] = xn[d] + t;
    }
    ln8(y, p.ln2_g, p.ln2_b);
    #pragma unroll
    for (int d = 0; d < 8; ++d) out[(size_t)pt*os + d] = y[d];
}

// ---------------------------------------------------------------------------
// 3) CNN encoder: conv(s=2,p=1)+leaky, batch-stats BN, pool, proj
// ---------------------------------------------------------------------------
__global__ void conv_leaky_kernel(const float* __restrict__ in,
                                  const float* __restrict__ w,
                                  const float* __restrict__ bias,
                                  float* __restrict__ out,
                                  int Cin, int Cout, int Hin, int Win,
                                  int Hout, int Wout) {
    int t = blockIdx.x * blockDim.x + threadIdx.x;
    int total = Bv * Cout * Hout * Wout;
    if (t >= total) return;
    int x = t % Wout, tt = t / Wout;
    int y = tt % Hout; tt /= Hout;
    int co = tt % Cout; int b = tt / Cout;
    float acc = bias[co];
    const float* wb = w + (size_t)co * Cin * 9;
    const float* ib = in + (size_t)b * Cin * Hin * Win;
    for (int ci = 0; ci < Cin; ++ci) {
        #pragma unroll
        for (int ky = 0; ky < 3; ++ky) {
            int iy = y * 2 + ky - 1;
            if (iy < 0 || iy >= Hin) continue;
            #pragma unroll
            for (int kx = 0; kx < 3; ++kx) {
                int ix = x * 2 + kx - 1;
                if (ix < 0 || ix >= Win) continue;
                acc += ib[((size_t)ci * Hin + iy) * Win + ix] * wb[ci * 9 + ky * 3 + kx];
            }
        }
    }
    out[t] = acc > 0.f ? acc : 0.2f * acc;
}

__global__ __launch_bounds__(256) void bn_stats_kernel(const float* __restrict__ x,
                                                       float* __restrict__ mean,
                                                       float* __restrict__ var,
                                                       int C, int HW) {
    int c = blockIdx.x;
    __shared__ float ssum[256], ssq[256];
    float s = 0.f, qq = 0.f;
    int n = Bv * HW;
    for (int i = threadIdx.x; i < n; i += 256) {
        int b = i / HW, sp = i % HW;
        float v = x[((size_t)b * C + c) * HW + sp];
        s += v; qq += v * v;
    }
    ssum[threadIdx.x] = s; ssq[threadIdx.x] = qq;
    __syncthreads();
    for (int off = 128; off > 0; off >>= 1) {
        if (threadIdx.x < off) {
            ssum[threadIdx.x] += ssum[threadIdx.x + off];
            ssq[threadIdx.x]  += ssq[threadIdx.x + off];
        }
        __syncthreads();
    }
    if (threadIdx.x == 0) {
        float m = ssum[0] / (float)n;
        mean[c] = m;
        var[c] = ssq[0] / (float)n - m * m;
    }
}

__global__ void bn_apply_kernel(float* __restrict__ x,
                                const float* __restrict__ mean,
                                const float* __restrict__ var,
                                const float* __restrict__ g,
                                const float* __restrict__ b, int C, int HW) {
    int t = blockIdx.x * blockDim.x + threadIdx.x;
    int total = Bv * C * HW;
    if (t >= total) return;
    int c = (t / HW) % C;
    x[t] = (x[t] - mean[c]) * rsqrtf(var[c] + 1e-5f) * g[c] + b[c];
}

__global__ void pool_kernel(const float* __restrict__ x, float* __restrict__ out,
                            int C, int HW) {
    int t = blockIdx.x * blockDim.x + threadIdx.x;
    if (t >= Bv * C) return;
    const float* p = x + (size_t)t * HW;
    float s = 0.f;
    for (int i = 0; i < HW; ++i) s += p[i];
    out[t] = s / (float)HW;
}

__global__ void proj_kernel(const float* __restrict__ pooled,
                            const float* __restrict__ w,
                            const float* __restrict__ b,
                            float* __restrict__ lat) {
    int t = blockIdx.x * blockDim.x + threadIdx.x;
    if (t >= Bv * HIDDEN) return;
    int bb = t / HIDDEN, o = t % HIDDEN;
    float acc = b[o];
    const float* pr = pooled + (size_t)bb * 256;
    const float* wr = w + (size_t)o * 256;
    for (int i = 0; i < 256; ++i) acc += pr[i] * wr[i];
    lat[t] = acc;
}

// ---------------------------------------------------------------------------
// 4) MLP head on the matrix pipe: v_wmma_f32_16x16x32_f16
// ---------------------------------------------------------------------------
__global__ void build_a0_kernel(const float* __restrict__ coords,
                                const float* __restrict__ attn_out,
                                const float* __restrict__ lat,
                                f16* __restrict__ a0) {
    int t = blockIdx.x * blockDim.x + threadIdx.x;
    if (t >= PTS * KP0) return;
    int p = t / KP0, k = t % KP0;
    int b = p >> 10;
    float v;
    if (k < 2)        v = coords[p * 2 + k];
    else if (k < 98)  v = attn_out[(size_t)p * 96 + (k - 2)];
    else if (k < 354) v = lat[(size_t)b * HIDDEN + (k - 98)];
    else              v = 0.f;
    a0[t] = (f16)v;
}

__global__ void cvt_pad_kernel(const float* __restrict__ src, f16* __restrict__ dst,
                               int rows, int cin, int cout) {
    int t = blockIdx.x * blockDim.x + threadIdx.x;
    if (t >= rows * cout) return;
    int r = t / cout, k = t % cout;
    dst[t] = (k < cin) ? (f16)src[(size_t)r * cin + k] : (f16)0.f;
}

// One 16x16 output tile per wave32; 8 waves / block. All 8 waves of a block
// share the same mtile (8 divides ntls=16), so the 16xKp A stripe is staged
// once into LDS via the CDNA5 async path (global_load_async_to_lds_b128,
// ASYNCcnt) and reused by all waves from ds_load. EXEC all-ones by
// construction (grid exactly covers tile space, uniform loops).
// A fragment (16x32 f16): lanes 0-15 -> K blocks {k..k+7, k+16..k+23},
// lanes 16-31 -> {k+8..k+15, k+24..k+31} (ISA 7.12.2 table).
// B fragment (32x16 f16): same K blocks as A, lanes 0-15 {k..k+7, k+16..k+23}, 16-31 {k+8..k+15, k+24..k+31}.
__global__ __launch_bounds__(256) void wmma_gemm_bias_gelu(
        const f16* __restrict__ A, const f16* __restrict__ W,
        const float* __restrict__ bias, f16* __restrict__ Out,
        int M, int N, int Kp, int apply_gelu) {
    __shared__ __align__(16) f16 atile[16 * MAXKP];
    const int lane  = threadIdx.x & 31;
    const int wave  = threadIdx.x >> 5;
    const int tile0 = blockIdx.x * 8;
    const int ntls  = N >> 4;
    const int mtile = tile0 / ntls;               // shared by all 8 waves
    const int ntile = (tile0 % ntls) + wave;
    const int r16   = lane & 15;
    const int grp   = lane >> 4;

    // ---- async-stage A stripe (16 rows x Kp halves) into LDS ----
    const int chunksPerRow = Kp >> 3;             // 16B chunks per row
    const int chunks = 16 * chunksPerRow;         // 768 (Kp=384) or 512 (Kp=256)
    for (int i = threadIdx.x; i < chunks; i += 256) {
        int row = i / chunksPerRow;
        int col = (i - row * chunksPerRow) << 3;
        const f16* g = A + (size_t)(mtile * 16 + row) * Kp + col;
        unsigned lds_off = (unsigned)(size_t)(atile + row * Kp + col);
        asm volatile("global_load_async_to_lds_b128 %0, %1, off"
                     :: "v"(lds_off), "v"((unsigned long long)(size_t)g)
                     : "memory");
    }
    asm volatile("s_wait_asynccnt 0" ::: "memory");
    __syncthreads();

    const f16* arow = atile + r16 * Kp;           // LDS source for A fragments
    const f16* wrow = W + (size_t)(ntile * 16 + r16) * Kp;

    v8f c = {0.f, 0.f, 0.f, 0.f, 0.f, 0.f, 0.f, 0.f};
    for (int k = 0; k < Kp; k += 32) {
        union { v16h v; v8h h[2]; } a, bm;
        const int ka0 = k + grp * 8;
        const int ka1 = k + 16 + grp * 8;
        a.h[0]  = *(const v8h*)(arow + ka0);
        a.h[1]  = *(const v8h*)(arow + ka1);
        const int kb0 = k + grp * 8;
        bm.h[0] = *(const v8h*)(wrow + kb0);
        bm.h[1] = *(const v8h*)(wrow + kb0 + 16);
        c = __builtin_amdgcn_wmma_f32_16x16x32_f16(
                false, a.v, false, bm.v, (short)0, c, false, false);
    }
    const int ncol = ntile * 16 + r16;
    const float bn = bias[ncol];
    #pragma unroll
    for (int r = 0; r < 8; ++r) {
        int m = mtile * 16 + r + grp * 8;   // C/D layout: lanes 0-15 M=r, 16-31 M=r+8
        float v = c[r] + bn;
        if (apply_gelu) v = gelu_exact(v);
        Out[(size_t)m * N + ncol] = (f16)v;
    }
}

__global__ void out_layer_kernel(const f16* __restrict__ A,
                                 const float* __restrict__ w,
                                 const float* __restrict__ b,
                                 float* __restrict__ out) {
    int t = blockIdx.x * blockDim.x + threadIdx.x;
    if (t >= PTS * 3) return;
    int p = t / 3, c = t % 3;
    float acc = b[c];
    const f16* ar = A + (size_t)p * HIDDEN;
    const float* wr = w + (size_t)c * HIDDEN;
    for (int i = 0; i < HIDDEN; ++i) acc += (float)ar[i] * wr[i];
    out[t] = 1.f / (1.f + expf(-acc));
}

// ---------------------------------------------------------------------------
// Host orchestration
// ---------------------------------------------------------------------------
// Input flattening assumption: top-level dict insertion order
// (lr_image, query_coords, params), params as sorted pytree:
//   attn (12 levels x 12 sorted tensors), lr (14), mlp (12), tables (12).
#define IDX_IMG     0
#define IDX_COORDS  1
#define ATTN_BASE   2     // + l*12 + {ff_b1,ff_b2,ff_w1,ff_w2,in_b,in_w,ln1_b,ln1_g,ln2_b,ln2_g,out_b,out_w}
#define LR_BASE     146   // bn0_b,bn0_g,bn1_b,bn1_g,bn2_b,bn2_g,conv0_b,conv0_w,conv1_b,conv1_w,conv2_b,conv2_w,proj_b,proj_w
#define MLP_BASE    160   // b0..b4, b_out, w0..w4, w_out
#define TAB_BASE    172   // 12 tables

extern "C" void kernel_launch(void* const* d_in, const int* in_sizes, int n_in,
                              void* d_out, int out_size, void* d_ws, size_t ws_size,
                              hipStream_t stream) {
    (void)in_sizes; (void)n_in; (void)out_size; (void)ws_size;
    auto F = [&](int i) { return (const float*)d_in[i]; };

    // ---- workspace bump allocator (256B aligned) ----
    char* wsp = (char*)d_ws;
    auto alloc = [&](size_t bytes) -> void* {
        void* p = (void*)wsp;
        wsp += (bytes + 255) & ~(size_t)255;
        return p;
    };
    float* feats    = (float*)alloc((size_t)NLEV * PTS * FPL * 4);
    float* qb       = (float*)alloc((size_t)PTS * FPL * 4);
    float* kb       = (float*)alloc((size_t)PTS * FPL * 4);
    float* vb       = (float*)alloc((size_t)PTS * FPL * 4);
    float* attn_out = (float*)alloc((size_t)PTS * 96 * 4);
    float* c0       = (float*)alloc((size_t)Bv * 64  * 32 * 32 * 4);
    float* c1       = (float*)alloc((size_t)Bv * 128 * 16 * 16 * 4);
    float* c2       = (float*)alloc((size_t)Bv * 256 * 8  * 8  * 4);
    float* bmean    = (float*)alloc(256 * 4);
    float* bvar     = (float*)alloc(256 * 4);
    float* pooled   = (float*)alloc((size_t)Bv * 256 * 4);
    float* lat      = (float*)alloc((size_t)Bv * HIDDEN * 4);
    f16*   a0       = (f16*)alloc((size_t)PTS * KP0 * 2);
    f16*   w0p      = (f16*)alloc((size_t)HIDDEN * KP0 * 2);
    f16*   w14p     = (f16*)alloc((size_t)4 * HIDDEN * HIDDEN * 2);
    f16*   actA     = (f16*)alloc((size_t)PTS * HIDDEN * 2);
    f16*   actB     = (f16*)alloc((size_t)PTS * HIDDEN * 2);

    // ---- grid config (matches Python _grid_config) ----
    HashArgs ha;
    {
        double s = std::exp((std::log(512.0) - std::log(16.0)) / (NLEV - 1));
        for (int l = 0; l < NLEV; ++l) {
            int r = (int)(16.0 * std::pow(s, (double)l));
            if (r < 2) r = 2;
            long long h = (long long)r * r;
            if (h > 131072LL) h = 131072LL;
            ha.res[l] = r;
            ha.hsz[l] = h;
            ha.tables[l] = F(TAB_BASE + l);
        }
    }

    const float* coords = F(IDX_COORDS);

    // 1) hash encode, all levels at once
    hash_encode_kernel<<<(NLEV * PTS + 255) / 256, 256, 0, stream>>>(coords, feats, ha);

    // 2) 12 sequential attention blocks
    for (int l = 0; l < NLEV; ++l) {
        int bidx = ATTN_BASE + l * 12;
        AttnP p;
        p.ff_b1 = F(bidx + 0);  p.ff_b2 = F(bidx + 1);
        p.ff_w1 = F(bidx + 2);  p.ff_w2 = F(bidx + 3);
        p.in_b  = F(bidx + 4);  p.in_w  = F(bidx + 5);
        p.ln1_b = F(bidx + 6);  p.ln1_g = F(bidx + 7);
        p.ln2_b = F(bidx + 8);  p.ln2_g = F(bidx + 9);
        p.out_b = F(bidx + 10); p.out_w = F(bidx + 11);
        const float* x = feats + (size_t)l * PTS * FPL;
        const float* ctx = (l == 0) ? x : (attn_out + (size_t)(l - 1) * FPL);
        int cs = (l == 0) ? FPL : 96;
        qkv_kernel<<<PTS / 256, 256, 0, stream>>>(x, FPL, ctx, cs, qb, kb, vb, p);
        attn_point_kernel<<<PTS / 256, 256, 0, stream>>>(
            x, FPL, qb, kb, vb, attn_out + (size_t)l * FPL, 96, p);
    }

    // 3) CNN encoder
    conv_leaky_kernel<<<(Bv*64*32*32 + 255)/256, 256, 0, stream>>>(
        F(IDX_IMG), F(LR_BASE + 7), F(LR_BASE + 6), c0, 3, 64, 64, 64, 32, 32);
    bn_stats_kernel<<<64, 256, 0, stream>>>(c0, bmean, bvar, 64, 32 * 32);
    bn_apply_kernel<<<(Bv*64*32*32 + 255)/256, 256, 0, stream>>>(
        c0, bmean, bvar, F(LR_BASE + 1), F(LR_BASE + 0), 64, 32 * 32);

    conv_leaky_kernel<<<(Bv*128*16*16 + 255)/256, 256, 0, stream>>>(
        c0, F(LR_BASE + 9), F(LR_BASE + 8), c1, 64, 128, 32, 32, 16, 16);
    bn_stats_kernel<<<128, 256, 0, stream>>>(c1, bmean, bvar, 128, 16 * 16);
    bn_apply_kernel<<<(Bv*128*16*16 + 255)/256, 256, 0, stream>>>(
        c1, bmean, bvar, F(LR_BASE + 3), F(LR_BASE + 2), 128, 16 * 16);

    conv_leaky_kernel<<<(Bv*256*8*8 + 255)/256, 256, 0, stream>>>(
        c1, F(LR_BASE + 11), F(LR_BASE + 10), c2, 128, 256, 16, 16, 8, 8);
    bn_stats_kernel<<<256, 256, 0, stream>>>(c2, bmean, bvar, 256, 8 * 8);
    bn_apply_kernel<<<(Bv*256*8*8 + 255)/256, 256, 0, stream>>>(
        c2, bmean, bvar, F(LR_BASE + 5), F(LR_BASE + 4), 256, 8 * 8);

    pool_kernel<<<(Bv*256 + 255)/256, 256, 0, stream>>>(c2, pooled, 256, 64);
    proj_kernel<<<(Bv*HIDDEN + 255)/256, 256, 0, stream>>>(
        pooled, F(LR_BASE + 13), F(LR_BASE + 12), lat);

    // 4) MLP head via WMMA
    build_a0_kernel<<<(PTS*KP0 + 255)/256, 256, 0, stream>>>(coords, attn_out, lat, a0);
    cvt_pad_kernel<<<(HIDDEN*KP0 + 255)/256, 256, 0, stream>>>(
        F(MLP_BASE + 6), w0p, HIDDEN, INDIM, KP0);
    for (int i = 0; i < 4; ++i)
        cvt_pad_kernel<<<(HIDDEN*HIDDEN + 255)/256, 256, 0, stream>>>(
            F(MLP_BASE + 7 + i), w14p + (size_t)i * HIDDEN * HIDDEN, HIDDEN, HIDDEN, HIDDEN);

    // layer 0: (2048 x 384) @ (384 x 256) -> 2048 tiles -> 256 blocks of 8 waves
    wmma_gemm_bias_gelu<<<(PTS/16)*(HIDDEN/16)/8, 256, 0, stream>>>(
        a0, w0p, F(MLP_BASE + 0), actA, PTS, HIDDEN, KP0, 1);
    f16* cur = actA; f16* nxt = actB;
    for (int i = 1; i < 5; ++i) {
        wmma_gemm_bias_gelu<<<(PTS/16)*(HIDDEN/16)/8, 256, 0, stream>>>(
            cur, w14p + (size_t)(i - 1) * HIDDEN * HIDDEN, F(MLP_BASE + i),
            nxt, PTS, HIDDEN, HIDDEN, 1);
        f16* t = cur; cur = nxt; nxt = t;
    }
    out_layer_kernel<<<(PTS*3 + 255)/256, 256, 0, stream>>>(
        cur, F(MLP_BASE + 11), F(MLP_BASE + 5), (float*)d_out);
}
